// MultiHeadGATLayer_41996190220976
// MI455X (gfx1250) — matrix operand order repaired and checked by reading, not compiled, not run on hardware
//
#include <hip/hip_runtime.h>
#include <hip/hip_bf16.h>

typedef __attribute__((ext_vector_type(16))) _Float16 v16h;
typedef __attribute__((ext_vector_type(8)))  float    v8f;

#define BB    8
#define NN    2048
#define FIN   64
#define HH    4
#define DD    32
#define FOUT  (HH * DD)      // 128
#define ALPHA 0.2f
#define TILE_I 16
#define TILE_J 32
#define NEG_BIG (-1.0e30f)

// ---------------------------------------------------------------------------
// Kernel 1: Wh = h @ W  (f32 math), emit Wh in f16 (WMMA operand precision)
// plus src[b,h,n] = Wh . a[h,:D], dst[b,h,n] = Wh . a[h,D:].
// One block per (b,n) row; 128 threads = 4 waves; wave w = head w, lane = d.
// ---------------------------------------------------------------------------
__global__ __launch_bounds__(128) void gat_proj_kernel(
    const float* __restrict__ hin,   // (B,N,FIN)
    const float* __restrict__ Wm,    // (FIN, FOUT)
    const float* __restrict__ am,    // (H, 2*DD)
    _Float16*   __restrict__ wh16,   // (B,H,N,DD) f16
    float*      __restrict__ srcv,   // (B,H,N)
    float*      __restrict__ dstv)   // (B,H,N)
{
    __shared__ float h_s[FIN];
    const int bn   = blockIdx.x;           // b*N + n
    const int b    = bn / NN;
    const int n    = bn % NN;
    const int tid  = threadIdx.x;          // output column c in [0,128)
    const int lane = tid & 31;             // d
    const int head = tid >> 5;             // h

    if (tid < FIN) h_s[tid] = hin[(size_t)bn * FIN + tid];
    __syncthreads();

    float acc = 0.0f;
#pragma unroll 8
    for (int k = 0; k < FIN; ++k)
        acc += h_s[k] * Wm[k * FOUT + tid];

    wh16[(((size_t)(b * HH + head)) * NN + n) * DD + lane] = (_Float16)acc;

    // per-head reductions for src/dst
    float sv = acc * am[head * (2 * DD) + lane];
    float dv = acc * am[head * (2 * DD) + DD + lane];
#pragma unroll
    for (int off = 16; off > 0; off >>= 1) {
        sv += __shfl_xor(sv, off, 32);
        dv += __shfl_xor(dv, off, 32);
    }
    if (lane == 0) {
        srcv[((size_t)(b * HH + head)) * NN + n] = sv;
        dstv[((size_t)(b * HH + head)) * NN + n] = dv;
    }
}

// ---------------------------------------------------------------------------
// Kernel 2: fused flash-style GAT attention -- barrier-free hot loop.
// Block = (b, 16-row i-tile); 4 waves = 4 heads. Each lane loads its own 16
// adjacency ints (A-fragment K-subset pattern, 4x global_load_b128); the 2KB
// adj tile is read once per block from HBM and re-served by the WGP cache to
// the sibling head-waves. All LDS state (dst/scale/l) is wave-private; wave
// DS ops are in-order, so only scheduling fences are needed -- no
// __syncthreads anywhere, waves pipeline their loads independently.
//
// A-fragment (f16 16x32, M=i, K=j) lane layout per ISA 7.12.2:
//   lanes 0-15 : M=lane,    elems 0..7 -> K=0..7,  elems 8..15 -> K=16..23
//   lanes16-31 : M=lane-16, elems 0..7 -> K=8..15, elems 8..15 -> K=24..31
// B-fragment (f16 32x16, K=j, N=d): lane L -> N=L&15, elem e -> same K as A elem e.
// C/D (f32 16x16): VGPR r, lanes 0-15 -> M=r, N=lane; lanes 16-31 -> M=r+8.
// ---------------------------------------------------------------------------
__global__ __launch_bounds__(128) void gat_attn_kernel(
    const int*      __restrict__ adj,    // (B,N,N)
    const _Float16* __restrict__ wh16,   // (B,H,N,DD)
    const float*    __restrict__ srcv,   // (B,H,N)
    const float*    __restrict__ dstv,   // (B,H,N)
    float*          __restrict__ out)    // (B,N,H*DD)
{
    __shared__ float dst_s[HH][TILE_J];       // wave-private (per head)
    __shared__ float scale_s[HH][TILE_I];     // wave-private (per head)
    __shared__ float l_s[HH][TILE_I];         // wave-private (per head)

    const int blk  = blockIdx.x;              // b*(N/16) + itile
    const int b    = blk / (NN / TILE_I);
    const int i0   = (blk % (NN / TILE_I)) * TILE_I;
    const int lane = threadIdx.x & 31;
    const int head = threadIdx.x >> 5;
    const int row  = lane & 15;               // M for the A-fragment
    const int grp  = lane >> 4;               // lane group (K subset select)

    const float*    dsth = dstv + ((size_t)(b * HH + head)) * NN;
    const _Float16* whh  = wh16 + ((size_t)(b * HH + head)) * NN * DD;
    // this lane's adjacency row, first K-run of its A-fragment subset
    const int* arow = adj + ((size_t)b) * NN * NN
                          + ((size_t)(i0 + row)) * NN + grp * 8;

    const float my_src = srcv[((size_t)(b * HH + head)) * NN + i0 + row];
    const float NEG_INF = -__builtin_inff();

    float m_run = NEG_BIG;
    float l_run = 0.0f;
    v8f acc_lo = {};
    v8f acc_hi = {};

#pragma unroll 2
    for (int j0 = 0; j0 < NN; j0 += TILE_J) {
        // ---- issue all global loads for this block up front (one region) --
        // B fragments (L2-resident Wh slice)
        const _Float16* bp = whh + (size_t)j0 * DD + (lane & 15);
        v16h blo, bhi;
#pragma unroll
        for (int e = 0; e < 16; ++e) {
            const int kb = grp * 8 + e + ((e >= 8) ? 8 : 0);
            blo[e] = bp[kb * DD];        // d = lane&15
            bhi[e] = bp[kb * DD + 16];   // d = 16 + (lane&15)
        }
        // adjacency: two contiguous 8-int runs = this lane's 16 A elements
        const int4 aa = *(const int4*)(arow + j0);        // K-sub 0..3
        const int4 ab = *(const int4*)(arow + j0 + 4);    // K-sub 4..7
        const int4 ac = *(const int4*)(arow + j0 + 16);   // K-sub 16..19
        const int4 ad = *(const int4*)(arow + j0 + 20);   // K-sub 20..23
        // dst slice -> wave-private LDS (cross-lane k access in exp loop)
        dst_s[head][lane] = dsth[j0 + lane];
        __builtin_amdgcn_wave_barrier();   // wave DS in-order: publish store

        if (j0 + 2 * TILE_J < NN)          // prefetch distance-2 adj chunk
            __builtin_prefetch(arow + j0 + 2 * TILE_J, 0, 1);

        const int avi[16] = { aa.x, aa.y, aa.z, aa.w,  ab.x, ab.y, ab.z, ab.w,
                              ac.x, ac.y, ac.z, ac.w,  ad.x, ad.y, ad.z, ad.w };

        // ---- logits for this lane's 16 A-fragment elements ----
        float ev[16];
        float lmax = NEG_INF;
#pragma unroll
        for (int e = 0; e < 16; ++e) {
            const int k = grp * 8 + e + ((e >= 8) ? 8 : 0);
            float x = my_src + dst_s[head][k];
            x = fmaxf(x, ALPHA * x);                 // leaky relu (alpha<1)
            ev[e] = (avi[e] != 0) ? x : NEG_INF;     // -inf => exp()==0 below
            lmax = fmaxf(lmax, ev[e]);
        }
        // row max across the (lane, lane^16) pair; floor keeps m finite
        const float rmax  = fmaxf(lmax, __shfl_xor(lmax, 16, 32));
        const float m_new = fmaxf(m_run, rmax);      // >= NEG_BIG always
        const float scale = __expf(m_run - m_new);

        // exponentiate straight into the A fragment (p in [0,1])
        float lsum = 0.0f;
        v16h afrag;
#pragma unroll
        for (int e = 0; e < 16; ++e) {
            const float p = __expf(ev[e] - m_new);   // masked -> exp(-inf)=0
            lsum += p;
            afrag[e] = (_Float16)p;
        }
        lsum += __shfl_xor(lsum, 16, 32);
        l_run = l_run * scale + lsum;
        m_run = m_new;

        // scale_s is wave-private; wave DS ops are in-order.
        if (lane < 16) scale_s[head][row] = scale;
        __builtin_amdgcn_wave_barrier();

        // rescale running accumulators (C layout: M = r + 8*grp)
#pragma unroll
        for (int r = 0; r < 8; ++r) {
            const float sc = scale_s[head][r + grp * 8];
            acc_lo[r] *= sc;
            acc_hi[r] *= sc;
        }

        acc_lo = __builtin_amdgcn_wmma_f32_16x16x32_f16(
            false, afrag, false, blo, (short)0, acc_lo, false, false);
        acc_hi = __builtin_amdgcn_wmma_f32_16x16x32_f16(
            false, afrag, false, bhi, (short)0, acc_hi, false, false);
    }

    // ---- epilogue: divide by softmax denominator, store (B,N,H*DD) ----
    if (lane < 16) l_s[head][row] = l_run;
    __builtin_amdgcn_wave_barrier();              // wave-private, in-order DS

    float* outp = out + ((size_t)b * NN + i0) * FOUT + head * DD;
    const int ncol = lane & 15;
#pragma unroll
    for (int r = 0; r < 8; ++r) {
        const int M = r + grp * 8;
        const float inv = 1.0f / l_s[head][M];
        outp[(size_t)M * FOUT + ncol]      = acc_lo[r] * inv;
        outp[(size_t)M * FOUT + 16 + ncol] = acc_hi[r] * inv;
    }
}

// ---------------------------------------------------------------------------
extern "C" void kernel_launch(void* const* d_in, const int* in_sizes, int n_in,
                              void* d_out, int out_size, void* d_ws, size_t ws_size,
                              hipStream_t stream) {
    const float* h_ptr   = (const float*)d_in[0];  // (B,N,FIN)
    const int*   adj_ptr = (const int*)d_in[1];    // (B,N,N)
    const float* W_ptr   = (const float*)d_in[2];  // (FIN,H*D)
    const float* a_ptr   = (const float*)d_in[3];  // (H,2D)
    float*       out_ptr = (float*)d_out;          // (B,N,H*D)

    // workspace layout
    const size_t wh16_elems = (size_t)BB * HH * NN * DD;        // 2M halfs = 4MB
    const size_t sd_elems   = (size_t)BB * HH * NN;             // 64K floats
    _Float16* wh16 = (_Float16*)d_ws;
    float* srcv = (float*)((char*)d_ws + wh16_elems * sizeof(_Float16));
    float* dstv = srcv + sd_elems;

    gat_proj_kernel<<<BB * NN, 128, 0, stream>>>(
        h_ptr, W_ptr, a_ptr, wh16, srcv, dstv);

    gat_attn_kernel<<<BB * (NN / TILE_I), 128, 0, stream>>>(
        adj_ptr, wh16, srcv, dstv, out_ptr);
}
